// DCRNN_21715354649731
// MI455X (gfx1250) — matrix operand order repaired and checked by reading, not compiled, not run on hardware
//
#include <hip/hip_runtime.h>
#include <cstdint>

// ---------------------------------------------------------------------------
// DCRNN cell for MI455X (gfx1250, wave32, WMMA).
//   Z = sigmoid(dconv(XH, Wz)), R = sigmoid(dconv(XH, Wr))
//   Ht = tanh(dconv([X | H*R], Wh)),  out = Z*H + (1-Z)*Ht
// Sparse Chebyshev propagations are weight-independent -> computed once for
// {Z,R} and once for H~.  Each dconv's 6 GEMMs collapse into one
// [N,480]x[480,64] bf16 WMMA GEMM (K blocks: XH,T1o,T1i,T2o,T2i).
// Both A and B operands are pre-swizzled into WMMA fragment layout (A by a
// bandwidth-bound pack pass, B at weight-prepack time), so the GEMM inner
// loop is pure vector loads + v_wmma_f32_16x16x32_bf16.
// ---------------------------------------------------------------------------

typedef __attribute__((ext_vector_type(16))) __bf16 v16bf;
typedef __attribute__((ext_vector_type(8)))  float  v8f;

#define F_IN   32
#define F_OUT  64
#define FBLK   96      // F_IN + F_OUT, one K-block
#define NBLKS  5       // XH, T1o, T1i, T2o, T2i
#define KSTEPS 3       // 96 / 32
#define NKS    (NBLKS * KSTEPS)       // 15 K-steps of 32
#define KTOT   480     // NBLKS * FBLK
#define NFRAGS (NKS * 4)              // 60 B-fragments, 512 ushorts each

__device__ __forceinline__ unsigned short f2bf_bits(float f) {
  unsigned u = __builtin_bit_cast(unsigned, f);
  u += 0x7FFFu + ((u >> 16) & 1u);           // round-to-nearest-even
  return (unsigned short)(u >> 16);
}

// ----------------------------- utility kernels -----------------------------

__global__ void zero_f32(float* __restrict__ p, long long n) {
  long long i = (long long)blockIdx.x * blockDim.x + threadIdx.x;
  if (i < n) p[i] = 0.f;
}

__global__ void degree_kernel(const long long* __restrict__ row,
                              const long long* __restrict__ col,
                              const float* __restrict__ ew,
                              float* __restrict__ deg_out,
                              float* __restrict__ deg_in, int E) {
  int e = blockIdx.x * blockDim.x + threadIdx.x;
  if (e >= E) return;
  float w = ew[e];
  atomicAdd(&deg_out[(int)row[e]], w);
  atomicAdd(&deg_in[(int)col[e]], w);
}

__global__ void build_xh(const float* __restrict__ X, const float* __restrict__ H,
                         float* __restrict__ XH, int N, int Npad) {
  int i = blockIdx.x * blockDim.x + threadIdx.x;
  if (i >= Npad * FBLK) return;
  int n = i / FBLK, c = i % FBLK;
  float v = 0.f;
  if (n < N) v = (c < F_IN) ? X[(size_t)n * F_IN + c] : H[(size_t)n * F_OUT + (c - F_IN)];
  XH[i] = v;
}

// XH[:, 32:96] = H * R   (X part untouched, pad rows untouched/zero)
__global__ void build_xhr(const float* __restrict__ H, const float* __restrict__ R,
                          float* __restrict__ XH, int N) {
  int i = blockIdx.x * blockDim.x + threadIdx.x;
  if (i >= N * F_OUT) return;
  int n = i / F_OUT, c = i % F_OUT;
  XH[(size_t)n * FBLK + F_IN + c] = H[i] * R[i];
}

// dst[d] += (ew[e]/deg[s]) * src[s]; one thread per (edge, 4-feature chunk)
__global__ void prop_kernel(const float* __restrict__ src,
                            const long long* __restrict__ sidx,
                            const long long* __restrict__ didx,
                            const float* __restrict__ ew,
                            const float* __restrict__ deg,
                            float* __restrict__ dst, int E) {
  int i = blockIdx.x * blockDim.x + threadIdx.x;
  if (i >= E * (FBLK / 4)) return;
  int e = i / (FBLK / 4);
  int c = (i % (FBLK / 4)) * 4;
  long long s = sidx[e], d = didx[e];
  float coeff = ew[e] / deg[s];
  const float* sp = src + (size_t)s * FBLK + c;
  float* dp = dst + (size_t)d * FBLK + c;
  float4 v = *(const float4*)sp;
  atomicAdd(dp + 0, coeff * v.x);
  atomicAdd(dp + 1, coeff * v.y);
  atomicAdd(dp + 2, coeff * v.z);
  atomicAdd(dp + 3, coeff * v.w);
}

// T2{o,i} = 2*P(T1{o,i}) - XH ; T2o,T2i are contiguous (2*perbuf floats)
__global__ void cheb_finalize(float* __restrict__ T2, const float* __restrict__ XH,
                              int perbuf) {
  int i = blockIdx.x * blockDim.x + threadIdx.x;
  if (i >= 2 * perbuf) return;
  int off = (i < perbuf) ? i : i - perbuf;
  T2[i] = 2.f * T2[i] - XH[off];
}

// Pack W[2,3,96,64] fp32 -> bf16 in WMMA B-fragment-swizzled order:
//   out[(ks*4+nt)*512 + lane*16 + e]
//   = Weff[row = (ks/3)*96 + (ks%3)*32 + (lane>=16?8:0) + (e<8 ? e : e+8), col = nt*16 + (lane&15)]
// where Weff block 0 = W[0,0]+W[1,0]; blocks 1..4 = W[0,1],W[1,1],W[0,2],W[1,2].
__global__ void prepack_kernel(const float* __restrict__ W, unsigned short* __restrict__ Wb) {
  int idx = blockIdx.x * blockDim.x + threadIdx.x;
  if (idx >= KTOT * F_OUT) return;
  int frag = idx >> 9;             // /512
  int lane = (idx >> 4) & 31;
  int e    = idx & 15;
  int b  = frag / (KSTEPS * 4);
  int s  = (frag % (KSTEPS * 4)) >> 2;
  int nt = frag & 3;
  int rr = s * 32 + ((lane >> 4) << 3) + (e < 8 ? e : e + 8);   // row within 96-block
  int j  = nt * 16 + (lane & 15);
  float v;
  if (b == 0) {
    v = W[(0 * 3 + 0) * FBLK * F_OUT + rr * F_OUT + j] +
        W[(1 * 3 + 0) * FBLK * F_OUT + rr * F_OUT + j];
  } else {
    int d = (b - 1) & 1;
    int k = 1 + ((b - 1) >> 1);
    v = W[((d * 3 + k) * FBLK + rr) * F_OUT + j];
  }
  Wb[idx] = f2bf_bits(v);
}

// Pack the 5 fp32 feature blocks into one bf16 A buffer in WMMA A-fragment
// layout: Apk[((t*NKS + ks)*32 + lane)*16 + e]
//   = blocks[ks/3][row = t*16 + (lane&15),
//                  k = (ks%3)*32 + (lane>>4)*8 + (e<8 ? e : e+8)]
// One thread handles one 8-element half (8 contiguous fp32 -> 16B bf16 store).
__global__ void pack_a_kernel(const float* B0, const float* B1, const float* B2,
                              const float* B3, const float* B4,
                              unsigned short* __restrict__ Apk, int nTiles) {
  int idx = blockIdx.x * blockDim.x + threadIdx.x;
  if (idx >= nTiles * NKS * 64) return;
  int half = idx & 1;
  int lane = (idx >> 1) & 31;
  int ks   = (idx >> 6) % NKS;
  int t    = (idx >> 6) / NKS;
  const float* blocks[NBLKS] = {B0, B1, B2, B3, B4};
  int b = ks / KSTEPS, s = ks % KSTEPS;
  int row = t * 16 + (lane & 15);
  int kb  = s * 32 + (lane >> 4) * 8 + (half ? 16 : 0);
  const float* src = blocks[b] + (size_t)row * FBLK + kb;
  float4 f0 = *(const float4*)(src);
  float4 f1 = *(const float4*)(src + 4);
  uint4 w;
  w.x = (unsigned)f2bf_bits(f0.x) | ((unsigned)f2bf_bits(f0.y) << 16);
  w.y = (unsigned)f2bf_bits(f0.z) | ((unsigned)f2bf_bits(f0.w) << 16);
  w.z = (unsigned)f2bf_bits(f1.x) | ((unsigned)f2bf_bits(f1.y) << 16);
  w.w = (unsigned)f2bf_bits(f1.z) | ((unsigned)f2bf_bits(f1.w) << 16);
  *(uint4*)(Apk + ((size_t)((t * NKS + ks) * 32 + lane) * 16) + half * 8) = w;
}

// ------------------------------- WMMA GEMM ---------------------------------
// One wave computes a 16x64 strip; A and B fragments are direct vector loads
// in fragment layout; fp32 accumulation via v_wmma_f32_16x16x32_bf16.

__device__ __forceinline__ void gemm_tile(const v16bf* __restrict__ Ap,
                                          const v16bf* __restrict__ ldsW,
                                          int tile, int lane, v8f acc[4]) {
  const v16bf* arow = Ap + (size_t)tile * NKS * 32 + lane;
  for (int ks = 0; ks < NKS; ++ks) {
    v16bf a = arow[ks * 32];                      // 32B coalesced global load
    if (ks + 1 < NKS) __builtin_prefetch(arow + (ks + 1) * 32, 0, 0);
#pragma unroll
    for (int nt = 0; nt < 4; ++nt) {
      v16bf bb = ldsW[(ks * 4 + nt) * 32 + lane]; // 32B LDS vector load
      acc[nt] = __builtin_amdgcn_wmma_f32_16x16x32_bf16(
          false, a, false, bb, (short)0, acc[nt], false, false);
    }
  }
}

#define STAGE_W(Wb)                                                         \
  __shared__ v16bf ldsW[NFRAGS * 32];                                       \
  {                                                                         \
    const uint4* wsrc = (const uint4*)(Wb);                                 \
    uint4* wdst = (uint4*)ldsW;                                             \
    for (int i = threadIdx.x; i < (KTOT * F_OUT) / 8; i += blockDim.x)      \
      wdst[i] = wsrc[i];                                                    \
  }                                                                         \
  __syncthreads();

__global__ __launch_bounds__(256) void gemm_sigmoid(
    const unsigned short* __restrict__ Apk,
    const unsigned short* __restrict__ Wb, const float* __restrict__ bias,
    float* __restrict__ Out, int N) {
  STAGE_W(Wb);
  int lane = threadIdx.x & 31;
  int wave = threadIdx.x >> 5;
  int tile = blockIdx.x * 8 + wave;
  int rowBase = tile * 16;
  v8f acc[4] = {};
  gemm_tile((const v16bf*)Apk, ldsW, tile, lane, acc);
  int laneHi = lane >> 4, nloc = lane & 15;
#pragma unroll
  for (int nt = 0; nt < 4; ++nt)
#pragma unroll
    for (int r = 0; r < 8; ++r) {
      int gr = rowBase + laneHi * 8 + r;
      int gc = nt * 16 + nloc;
      if (gr < N) {
        float v = acc[nt][r] + bias[gc];
        Out[(size_t)gr * F_OUT + gc] = 1.f / (1.f + __expf(-v));
      }
    }
}

__global__ __launch_bounds__(256) void gemm_final(
    const unsigned short* __restrict__ Apk,
    const unsigned short* __restrict__ Wb, const float* __restrict__ bias,
    const float* __restrict__ Z, const float* __restrict__ Hin,
    float* __restrict__ Out, int N) {
  STAGE_W(Wb);
  int lane = threadIdx.x & 31;
  int wave = threadIdx.x >> 5;
  int tile = blockIdx.x * 8 + wave;
  int rowBase = tile * 16;
  v8f acc[4] = {};
  gemm_tile((const v16bf*)Apk, ldsW, tile, lane, acc);
  int laneHi = lane >> 4, nloc = lane & 15;
#pragma unroll
  for (int nt = 0; nt < 4; ++nt)
#pragma unroll
    for (int r = 0; r < 8; ++r) {
      int gr = rowBase + laneHi * 8 + r;
      int gc = nt * 16 + nloc;
      if (gr < N) {
        size_t o = (size_t)gr * F_OUT + gc;
        float g = tanhf(acc[nt][r] + bias[gc]);
        float z = Z[o];
        Out[o] = z * Hin[o] + (1.f - z) * g;
      }
    }
}

// ------------------------------- launcher ----------------------------------

extern "C" void kernel_launch(void* const* d_in, const int* in_sizes, int n_in,
                              void* d_out, int out_size, void* d_ws, size_t ws_size,
                              hipStream_t stream) {
  const float*     X  = (const float*)d_in[0];
  const long long* ei = (const long long*)d_in[1];   // int64 [2,E]
  const float*     ew = (const float*)d_in[2];
  const float*     H  = (const float*)d_in[3];
  const float*     Wz = (const float*)d_in[4];
  const float*     bz = (const float*)d_in[5];
  const float*     Wr = (const float*)d_in[6];
  const float*     br = (const float*)d_in[7];
  const float*     Wh = (const float*)d_in[8];
  const float*     bh = (const float*)d_in[9];

  int N = in_sizes[0] / F_IN;
  int E = in_sizes[2];
  int Npad = (N + 127) & ~127;
  int nTiles = Npad / 16;
  const long long* row = ei;
  const long long* col = ei + E;

  char* ws = (char*)d_ws;
  size_t off = 0;
  auto alloc = [&](size_t bytes) -> char* {
    char* p = ws + off;
    off = (off + bytes + 255) & ~(size_t)255;
    return p;
  };
  size_t per = (size_t)Npad * FBLK;                    // floats per feature block
  float* deg_out = (float*)alloc(sizeof(float) * Npad);
  float* deg_in  = (float*)alloc(sizeof(float) * Npad);    // contiguous with deg_out
  float* XH  = (float*)alloc(sizeof(float) * per);
  float* T1o = (float*)alloc(sizeof(float) * per);         // T1o..T2i contiguous
  float* T1i = (float*)alloc(sizeof(float) * per);
  float* T2o = (float*)alloc(sizeof(float) * per);
  float* T2i = (float*)alloc(sizeof(float) * per);
  float* Zb  = (float*)alloc(sizeof(float) * (size_t)Npad * F_OUT);
  float* Rb  = (float*)alloc(sizeof(float) * (size_t)Npad * F_OUT);
  unsigned short* Apk = (unsigned short*)alloc(sizeof(unsigned short) * (size_t)Npad * KTOT);
  unsigned short* Wbz = (unsigned short*)alloc(sizeof(unsigned short) * KTOT * F_OUT);
  unsigned short* Wbr = (unsigned short*)alloc(sizeof(unsigned short) * KTOT * F_OUT);
  unsigned short* Wbh = (unsigned short*)alloc(sizeof(unsigned short) * KTOT * F_OUT);
  (void)ws_size; (void)n_in; (void)out_size;

  dim3 blk(256);
  auto g1 = [](long long n) { return dim3((unsigned)((n + 255) / 256)); };
  long long pth = (long long)E * (FBLK / 4);
  long long packth = (long long)nTiles * NKS * 64;

  // weight prepack (independent of everything else)
  prepack_kernel<<<g1(KTOT * F_OUT), blk, 0, stream>>>(Wz, Wbz);
  prepack_kernel<<<g1(KTOT * F_OUT), blk, 0, stream>>>(Wr, Wbr);
  prepack_kernel<<<g1(KTOT * F_OUT), blk, 0, stream>>>(Wh, Wbh);

  // degrees
  zero_f32<<<g1(2LL * Npad), blk, 0, stream>>>(deg_out, 2LL * Npad);
  degree_kernel<<<g1(E), blk, 0, stream>>>(row, col, ew, deg_out, deg_in, E);

  // XH = [X | H] (padded rows zero)
  build_xh<<<g1((long long)per), blk, 0, stream>>>(X, H, XH, N, Npad);

  // Chebyshev propagations on XH (shared by Z and R gates)
  zero_f32<<<g1(4LL * per), blk, 0, stream>>>(T1o, 4LL * per);
  prop_kernel<<<g1(pth), blk, 0, stream>>>(XH,  row, col, ew, deg_out, T1o, E);
  prop_kernel<<<g1(pth), blk, 0, stream>>>(XH,  col, row, ew, deg_in,  T1i, E);
  prop_kernel<<<g1(pth), blk, 0, stream>>>(T1o, row, col, ew, deg_out, T2o, E);
  prop_kernel<<<g1(pth), blk, 0, stream>>>(T1i, col, row, ew, deg_in,  T2i, E);
  cheb_finalize<<<g1(2LL * per), blk, 0, stream>>>(T2o, XH, (int)per);

  // pack A (fp32 blocks -> bf16 fragment layout), then Z and R gate GEMMs
  pack_a_kernel<<<g1(packth), blk, 0, stream>>>(XH, T1o, T1i, T2o, T2i, Apk, nTiles);
  dim3 ggrid(Npad / 128);
  gemm_sigmoid<<<ggrid, blk, 0, stream>>>(Apk, Wbz, bz, Zb, N);
  gemm_sigmoid<<<ggrid, blk, 0, stream>>>(Apk, Wbr, br, Rb, N);

  // XHR = [X | H*R], re-run propagations (reuse T buffers)
  build_xhr<<<g1((long long)N * F_OUT), blk, 0, stream>>>(H, Rb, XH, N);
  zero_f32<<<g1(4LL * per), blk, 0, stream>>>(T1o, 4LL * per);
  prop_kernel<<<g1(pth), blk, 0, stream>>>(XH,  row, col, ew, deg_out, T1o, E);
  prop_kernel<<<g1(pth), blk, 0, stream>>>(XH,  col, row, ew, deg_in,  T1i, E);
  prop_kernel<<<g1(pth), blk, 0, stream>>>(T1o, row, col, ew, deg_out, T2o, E);
  prop_kernel<<<g1(pth), blk, 0, stream>>>(T1i, col, row, ew, deg_in,  T2i, E);
  cheb_finalize<<<g1(2LL * per), blk, 0, stream>>>(T2o, XH, (int)per);

  // pack A for H~ phase, then final GEMM + tanh + GRU gating -> output
  pack_a_kernel<<<g1(packth), blk, 0, stream>>>(XH, T1o, T1i, T2o, T2i, Apk, nTiles);
  gemm_final<<<ggrid, blk, 0, stream>>>(Apk, Wbh, bh, Zb, H, (float*)d_out, N);
}
